// Cosal_Module_5068061409754
// MI455X (gfx1250) — matrix operand order repaired and checked by reading, not compiled, not run on hardware
//
#include <hip/hip_runtime.h>
#include <hip/hip_bf16.h>
#include <math.h>

// ---------------------------------------------------------------------------
// CDNA5 (gfx1250), wave32. All GEMM-shaped stages use v_wmma_f32_16x16x32_f16
// with K-contiguous operands: A = weights/proxies (M = out-channel),
// B = activations in pixel-major [n][p][c] layout (N = pixel).
// Conv kernels compute 32x32 output tiles per wave (2x2 WMMA tiles) so each
// A/B fragment feeds two WMMAs: 2.0 b128-loads per WMMA issue.
// ---------------------------------------------------------------------------

typedef _Float16 f16;
typedef __attribute__((ext_vector_type(16))) _Float16 v16h;
typedef __attribute__((ext_vector_type(8)))  _Float16 v8h;
typedef __attribute__((ext_vector_type(8)))  float    v8f;

#define HW   4096
#define NIMG 64
#define CIN  256

// ---- WMMA fragment helpers (ISA 7.12.2, wave32) ---------------------------
__device__ __forceinline__ v16h load_a_frag(const f16* row, int kb, int hi) {
    return *(const v16h*)(row + kb + 16 * hi);
}
__device__ __forceinline__ v16h load_b_frag(const f16* row, int kb, int hi) {
    return *(const v16h*)(row + kb + 16 * hi);
}
// C/D (16x16 f32): lane -> col N=lane&15 (pixel); vgpr v -> row M = v+8*(lane>>4)

__device__ __forceinline__ v8f wmma_f16(v16h a, v16h b, v8f c) {
    return __builtin_amdgcn_wmma_f32_16x16x32_f16(false, a, false, b,
                                                  (short)0, c, false, false);
}

__device__ __forceinline__ float wave_red_sum(float v) {
    #pragma unroll
    for (int off = 16; off; off >>= 1) v += __shfl_xor(v, off, 32);
    return v;
}

// ---------------------------------------------------------------------------
__global__ void k_f2h(const float* __restrict__ src, f16* __restrict__ dst, int n) {
    int i = blockIdx.x * blockDim.x + threadIdx.x;
    if (i < n) dst[i] = (f16)src[i];
}
// 3x3 conv weights: [o][c][tap] f32 -> [tap][o][c] f16
__global__ void k_w3(const float* __restrict__ src, f16* __restrict__ dst) {
    int i = blockIdx.x * blockDim.x + threadIdx.x;   // 128*128*9
    if (i >= 128 * 128 * 9) return;
    int tap = i % 9, c = (i / 9) % 128, o = i / (9 * 128);
    dst[((size_t)tap * 128 + o) * 128 + c] = (f16)src[i];
}

// ---------------------------------------------------------------------------
// NFs = feats / max(||feats||_c, eps): [n][c][p] f32 -> pixel-major [n][p][c] f16
__global__ void k_nfs(const float* __restrict__ feats, f16* __restrict__ nfs) {
    int tid = blockIdx.x * blockDim.x + threadIdx.x;   // 64*4096
    int n = tid >> 12, p = tid & 4095;
    const float* base = feats + (size_t)n * CIN * HW + p;
    float ss = 0.f;
    for (int c = 0; c < CIN; ++c) { float x = base[(size_t)c * HW]; ss += x * x; }
    float inv = 1.f / fmaxf(sqrtf(ss), 1e-12f);
    f16* ob = nfs + ((size_t)n * HW + p) * CIN;
    for (int c8 = 0; c8 < CIN / 8; ++c8) {
        v8h r;
        #pragma unroll
        for (int j = 0; j < 8; ++j)
            r[j] = (f16)(base[(size_t)(c8 * 8 + j) * HW] * inv);
        *(v8h*)(ob + c8 * 8) = r;
    }
}

// ---------------------------------------------------------------------------
// masked[v][n][c] = mean_p NFs*sism (v=0) / NFs*(1-sism) (v=1). block per n.
__global__ void k_masked(const f16* __restrict__ nfs, const float* __restrict__ sism,
                         float* __restrict__ masked) {
    int n = blockIdx.x, c = threadIdx.x;               // 256 threads
    const f16*  nb = nfs + (size_t)n * HW * CIN + c;
    const float* sb = sism + (size_t)n * HW;
    float sf = 0.f, st = 0.f;
    for (int p = 0; p < HW; ++p) {
        float v = (float)nb[(size_t)p * CIN];
        sf += v * sb[p];
        st += v;
    }
    masked[(size_t)n * 256 + c]              = sf * (1.f / HW);
    masked[(size_t)NIMG * 256 + n * 256 + c] = (st - sf) * (1.f / HW);
}

// ---------------------------------------------------------------------------
// co_proxy[n] = l2n(masked[n]); r[g] = l2n(mean_n masked). block per group.
__global__ void k_proxy(const float* __restrict__ masked, f16* __restrict__ proxy16,
                        float* __restrict__ rvec) {
    int g = blockIdx.x, c = threadIdx.x;
    __shared__ float red[256];
    float rsum = 0.f;
    for (int n = 0; n < 16; ++n) {
        float v = masked[(size_t)(g * 16 + n) * 256 + c];
        rsum += v;
        red[c] = v * v; __syncthreads();
        for (int s = 128; s; s >>= 1) { if (c < s) red[c] += red[c + s]; __syncthreads(); }
        float inv = 1.f / fmaxf(sqrtf(red[0]), 1e-12f);
        __syncthreads();
        proxy16[(size_t)(g * 16 + n) * 256 + c] = (f16)(v * inv);
    }
    float rm = rsum * (1.f / 16.f);
    red[c] = rm * rm; __syncthreads();
    for (int s = 128; s; s >>= 1) { if (c < s) red[c] += red[c + s]; __syncthreads(); }
    float inv = 1.f / fmaxf(sqrtf(red[0]), 1e-12f);
    rvec[g * 256 + c] = rm * inv;
}

// ---------------------------------------------------------------------------
// corr[n][p][m] = NFs[n,p,:].co_proxy[m,:]  — A=proxies(M=m), B=pixels(N=p)
__global__ void k_corr(const f16* __restrict__ nfs, const f16* __restrict__ proxy16,
                       float* __restrict__ corr) {
    int wid  = (blockIdx.x * blockDim.x + threadIdx.x) >> 5;  // 64*256 waves
    int lane = threadIdx.x & 31, hi = lane >> 4, lm = lane & 15;
    int n = wid >> 8, pt = wid & 255;
    int p0 = pt * 16, g = n >> 4;
    const f16* arow = proxy16 + (size_t)(g * 16 + lm) * 256;
    const f16* brow = nfs + ((size_t)n * HW + p0 + lm) * CIN;
    v8f acc = {};
    #pragma unroll
    for (int kb = 0; kb < CIN; kb += 32)
        acc = wmma_f16(load_a_frag(arow, kb, hi), load_b_frag(brow, kb, hi), acc);
    float* ob = corr + ((size_t)n * HW + p0 + lm) * 16 + 8 * hi;
    #pragma unroll
    for (int v = 0; v < 8; ++v) ob[v] = acc[v];
}

// 1/||corr[n][:, m]||  (one wave per (n,m))
__global__ void k_cnorm(const float* __restrict__ corr, float* __restrict__ cnorm) {
    int wid  = (blockIdx.x * blockDim.x + threadIdx.x) >> 5;  // 1024 waves
    int lane = threadIdx.x & 31;
    int n = wid >> 4, m = wid & 15;
    const float* cb = corr + (size_t)n * HW * 16 + m;
    float ss = 0.f;
    for (int p = lane; p < HW; p += 32) { float v = cb[(size_t)p * 16]; ss += v * v; }
    ss = wave_red_sum(ss);
    if (lane == 0) cnorm[wid] = 1.f / fmaxf(sqrtf(ss), 1e-12f);
}

// co_att[n][p] = minmax-norm( sum_m corr_hat )  (one block per n)
__global__ void k_coatt(const float* __restrict__ corr, const float* __restrict__ cnorm,
                        float* __restrict__ coatt) {
    int n = blockIdx.x, t = threadIdx.x;
    __shared__ float smn[256], smx[256], sc[16];
    if (t < 16) sc[t] = cnorm[n * 16 + t];
    __syncthreads();
    float lmn = 1e30f, lmx = -1e30f;
    for (int p = t; p < HW; p += 256) {
        const float* cp = corr + ((size_t)n * HW + p) * 16;
        float s = 0.f;
        #pragma unroll
        for (int m = 0; m < 16; ++m) s += cp[m] * sc[m];
        coatt[(size_t)n * HW + p] = s;
        lmn = fminf(lmn, s); lmx = fmaxf(lmx, s);
    }
    smn[t] = lmn; smx[t] = lmx; __syncthreads();
    for (int s = 128; s; s >>= 1) {
        if (t < s) { smn[t] = fminf(smn[t], smn[t + s]); smx[t] = fmaxf(smx[t], smx[t + s]); }
        __syncthreads();
    }
    float mn = smn[0], inv = 1.f / (smx[0] - smn[0] + 1e-12f);
    for (int p = t; p < HW; p += 256)
        coatt[(size_t)n * HW + p] = (coatt[(size_t)n * HW + p] - mn) * inv;
}

// ---------------------------------------------------------------------------
// score[g][i] = pix . r
__global__ void k_score(const f16* __restrict__ nfs, const float* __restrict__ rvec,
                        float* __restrict__ score) {
    int tid = blockIdx.x * blockDim.x + threadIdx.x;   // 64*4096
    int n = tid >> 12, p = tid & 4095, g = n >> 4;
    const f16* nb = nfs + ((size_t)n * HW + p) * CIN;
    const float* r = rvec + g * 256;
    float s = 0.f;
    for (int c8 = 0; c8 < CIN / 8; ++c8) {
        v8h h = *(const v8h*)(nb + c8 * 8);
        #pragma unroll
        for (int j = 0; j < 8; ++j) s += (float)h[j] * r[c8 * 8 + j];
    }
    score[(size_t)g * 65536 + (size_t)(n & 15) * HW + p] = s;
}

// top-32 argmax rounds + gather co_rep (one block per group)
__global__ void k_topk(float* __restrict__ score, const f16* __restrict__ nfs,
                       f16* __restrict__ corep) {
    int g = blockIdx.x, t = threadIdx.x;
    __shared__ float sv[256]; __shared__ int si[256]; __shared__ int sel[32];
    for (int k = 0; k < 32; ++k) {
        float bv = -1e30f; int bi = 0;
        for (int i = t; i < 65536; i += 256) {
            float v = score[(size_t)g * 65536 + i];
            if (v > bv) { bv = v; bi = i; }
        }
        sv[t] = bv; si[t] = bi; __syncthreads();
        for (int s = 128; s; s >>= 1) {
            if (t < s && sv[t + s] > sv[t]) { sv[t] = sv[t + s]; si[t] = si[t + s]; }
            __syncthreads();
        }
        if (t == 0) { sel[k] = si[0]; score[(size_t)g * 65536 + si[0]] = -1e30f; }
        __syncthreads();
    }
    for (int k = 0; k < 32; ++k) {
        int idx = sel[k];
        int n = g * 16 + (idx >> 12), p = idx & 4095;
        corep[((size_t)g * 32 + k) * 256 + t] = nfs[((size_t)n * HW + p) * CIN + t];
    }
}

// ---------------------------------------------------------------------------
// SUBX[n][p][k32] = (NFs . co_rep[k]) * co_att — A=co_rep(M=k), B=pixels
__global__ void k_cofs(const f16* __restrict__ nfs, const f16* __restrict__ corep,
                       const float* __restrict__ coatt, f16* __restrict__ subx) {
    int wid  = (blockIdx.x * blockDim.x + threadIdx.x) >> 5;  // 64*256*2 waves
    int lane = threadIdx.x & 31, hi = lane >> 4, lm = lane & 15;
    int n = wid >> 9, pt = (wid >> 1) & 255, kh = wid & 1;
    int p0 = pt * 16, g = n >> 4;
    const f16* arow = corep + ((size_t)g * 32 + kh * 16 + lm) * 256;
    const f16* brow = nfs + ((size_t)n * HW + p0 + lm) * CIN;
    v8f acc = {};
    #pragma unroll
    for (int kb = 0; kb < CIN; kb += 32)
        acc = wmma_f16(load_a_frag(arow, kb, hi), load_b_frag(brow, kb, hi), acc);
    float att = coatt[(size_t)n * HW + p0 + lm];
    v8h r;
    #pragma unroll
    for (int v = 0; v < 8; ++v) r[v] = (f16)(acc[v] * att);
    *(v8h*)(subx + ((size_t)n * HW + p0 + lm) * 32 + kh * 16 + 8 * hi) = r;
}

// ---------------------------------------------------------------------------
// 1x1 conv, 32x32 tile per wave: out[n][p][obase+o] = sum_c w[o][c]*in[n][p][c]+b[o]
template <int CI>
__global__ void k_conv1x1(const f16* __restrict__ in, const f16* __restrict__ w,
                          const float* __restrict__ bias, f16* __restrict__ out,
                          int ostr, int obase) {
    int wid  = (blockIdx.x * blockDim.x + threadIdx.x) >> 5;  // 64*128*4 waves
    int lane = threadIdx.x & 31, hi = lane >> 4, lm = lane & 15;
    int otp = wid & 3, pt = (wid >> 2) & 127, n = wid >> 9;
    int p0 = pt * 32, o0 = otp * 32;
    const f16* a0row = w + (size_t)(o0 + lm) * CI;
    const f16* a1row = a0row + (size_t)16 * CI;
    const f16* b0row = in + ((size_t)n * HW + p0 + lm) * CI;
    const f16* b1row = b0row + (size_t)16 * CI;
    v8f acc00 = {}, acc01 = {}, acc10 = {}, acc11 = {};
    #pragma unroll
    for (int kb = 0; kb < CI; kb += 32) {
        v16h a0 = load_a_frag(a0row, kb, hi), a1 = load_a_frag(a1row, kb, hi);
        v16h b0 = load_b_frag(b0row, kb, hi), b1 = load_b_frag(b1row, kb, hi);
        acc00 = wmma_f16(a0, b0, acc00); acc01 = wmma_f16(a0, b1, acc01);
        acc10 = wmma_f16(a1, b0, acc10); acc11 = wmma_f16(a1, b1, acc11);
    }
    #pragma unroll
    for (int t = 0; t < 2; ++t)
        #pragma unroll
        for (int j = 0; j < 2; ++j) {
            v8f acc = t ? (j ? acc11 : acc10) : (j ? acc01 : acc00);
            int o = o0 + t * 16 + 8 * hi;
            int p = p0 + j * 16 + lm;
            v8h r;
            #pragma unroll
            for (int v = 0; v < 8; ++v) r[v] = (f16)(acc[v] + bias[o + v]);
            *(v8h*)(out + ((size_t)n * HW + p) * ostr + obase + o) = r;
        }
}

// ---------------------------------------------------------------------------
// 3x3 conv 128->128, implicit GEMM, 32 out-channels x 32 pixels per wave
// (2x2 WMMA tiles; A and B fragments each feed two WMMAs).
// MODE 0: out16 = relu(gamma*(acc+b)/sqrt(1+eps) + beta)
// MODE 1: out16 = relu(skip + acc + b)
// MODE 2: out32[NCHW] = relu(skip + acc + b)   (final fp32 output)
template <int MODE>
__global__ void k_conv3x3(const f16* __restrict__ in, const f16* __restrict__ w3,
                          const float* __restrict__ bias, const float* __restrict__ gamma,
                          const float* __restrict__ beta, const f16* __restrict__ skip,
                          f16* __restrict__ out16, float* __restrict__ out32,
                          int ostr, int obase) {
    int wid  = (blockIdx.x * blockDim.x + threadIdx.x) >> 5;  // 64*128*4 waves
    int lane = threadIdx.x & 31, hi = lane >> 4, lm = lane & 15;
    int otp = wid & 3, pt = (wid >> 2) & 127, n = wid >> 9;
    int p0 = pt * 32, o0 = otp * 32;
    int y = p0 >> 6, xb = p0 & 63;
    const f16* inb = in + (size_t)n * HW * 128;
    const v16h bzero = {};
    v8f acc00 = {}, acc01 = {}, acc10 = {}, acc11 = {};
    #pragma unroll
    for (int tap = 0; tap < 9; ++tap) {
        int dy = tap / 3 - 1, dx = tap % 3 - 1;
        int iy = y + dy;
        bool rowok = (iy >= 0) & (iy < 64);
        int ix0 = xb + lm + dx;
        int ix1 = ix0 + 16;
        bool ok0 = rowok & (ix0 >= 0) & (ix0 < 64);
        bool ok1 = rowok & (ix1 >= 0) & (ix1 < 64);
        const f16* b0row = inb + (size_t)(iy * 64 + ix0) * 128;
        const f16* b1row = inb + (size_t)(iy * 64 + ix1) * 128;
        const f16* a0row = w3 + ((size_t)tap * 128 + o0 + lm) * 128;
        const f16* a1row = a0row + 16 * 128;
        if (ok0) __builtin_prefetch(b0row, 0, 3);
        #pragma unroll
        for (int kb = 0; kb < 128; kb += 32) {
            v16h a0 = load_a_frag(a0row, kb, hi), a1 = load_a_frag(a1row, kb, hi);
            v16h b0 = ok0 ? load_b_frag(b0row, kb, hi) : bzero;
            v16h b1 = ok1 ? load_b_frag(b1row, kb, hi) : bzero;
            acc00 = wmma_f16(a0, b0, acc00); acc01 = wmma_f16(a0, b1, acc01);
            acc10 = wmma_f16(a1, b0, acc10); acc11 = wmma_f16(a1, b1, acc11);
        }
    }
    #pragma unroll
    for (int t = 0; t < 2; ++t)
        #pragma unroll
        for (int j = 0; j < 2; ++j) {
            v8f acc = t ? (j ? acc11 : acc10) : (j ? acc01 : acc00);
            int o = o0 + t * 16 + 8 * hi;
            int p = p0 + j * 16 + lm;
            if (MODE == 0) {
                const float inv_bn = 0.9999950000374997f;   // 1/sqrt(1+1e-5)
                v8h r;
                #pragma unroll
                for (int v = 0; v < 8; ++v)
                    r[v] = (f16)fmaxf((acc[v] + bias[o + v]) * gamma[o + v] * inv_bn
                                      + beta[o + v], 0.f);
                *(v8h*)(out16 + ((size_t)n * HW + p) * ostr + obase + o) = r;
            } else if (MODE == 1) {
                v8h sk = *(const v8h*)(skip + ((size_t)n * HW + p) * 128 + o);
                v8h r;
                #pragma unroll
                for (int v = 0; v < 8; ++v)
                    r[v] = (f16)fmaxf((float)sk[v] + acc[v] + bias[o + v], 0.f);
                *(v8h*)(out16 + ((size_t)n * HW + p) * ostr + obase + o) = r;
            } else {
                v8h sk = *(const v8h*)(skip + ((size_t)n * HW + p) * 128 + o);
                #pragma unroll
                for (int v = 0; v < 8; ++v)
                    out32[((size_t)n * 128 + o + v) * HW + p] =
                        fmaxf((float)sk[v] + acc[v] + bias[o + v], 0.f);
            }
        }
}

// ---------------------------------------------------------------------------
extern "C" void kernel_launch(void* const* d_in, const int* in_sizes, int n_in,
                              void* d_out, int out_size, void* d_ws, size_t ws_size,
                              hipStream_t stream) {
    const float* feats = (const float*)d_in[0];
    const float* sism  = (const float*)d_in[1];
    const float* sw0 = (const float*)d_in[2];  const float* sb0  = (const float*)d_in[3];
    const float* sw1 = (const float*)d_in[4];  const float* sb1  = (const float*)d_in[5];
    const float* sg1 = (const float*)d_in[6];  const float* sbt1 = (const float*)d_in[7];
    const float* sw2 = (const float*)d_in[8];  const float* sb2  = (const float*)d_in[9];
    const float* ow0 = (const float*)d_in[10]; const float* ob0  = (const float*)d_in[11];
    const float* ow1 = (const float*)d_in[12]; const float* ob1  = (const float*)d_in[13];
    const float* og1 = (const float*)d_in[14]; const float* obt1 = (const float*)d_in[15];
    const float* ow2 = (const float*)d_in[16]; const float* ob2  = (const float*)d_in[17];
    float* out = (float*)d_out;

    char* ws = (char*)d_ws;
    size_t off = 0;
    auto alloc = [&](size_t bytes) -> void* {
        void* p = ws + off;
        off = (off + bytes + 255) & ~(size_t)255;
        return p;
    };
    f16*   NFS    = (f16*)  alloc((size_t)NIMG * HW * CIN * 2);        // 128 MB
    f16*   SW0F   = (f16*)  alloc(128 * 32 * 2);
    f16*   SW1F   = (f16*)  alloc(128 * 128 * 9 * 2);
    f16*   SW2F   = (f16*)  alloc(128 * 128 * 9 * 2);
    f16*   OW0F   = (f16*)  alloc(128 * 256 * 2);
    f16*   OW1F   = (f16*)  alloc(128 * 128 * 9 * 2);
    f16*   OW2F   = (f16*)  alloc(128 * 128 * 9 * 2);
    float* MASKED = (float*)alloc(2 * NIMG * 256 * 4);
    f16*   PROXY  = (f16*)  alloc(NIMG * 256 * 2);
    float* RVEC   = (float*)alloc(4 * 256 * 4);
    float* CORR   = (float*)alloc((size_t)NIMG * HW * 16 * 4);          // 16 MB
    float* CNORM  = (float*)alloc(NIMG * 16 * 4);
    float* COATT  = (float*)alloc((size_t)NIMG * HW * 4);
    float* SCORE  = (float*)alloc((size_t)4 * 65536 * 4);
    f16*   COREP  = (f16*)  alloc(4 * 32 * 256 * 2);
    f16*   SUBX   = (f16*)  alloc((size_t)NIMG * HW * 32 * 2);          // 16 MB
    f16*   Y0     = (f16*)  alloc((size_t)NIMG * HW * 128 * 2);         // 64 MB
    f16*   H1     = (f16*)  alloc((size_t)NIMG * HW * 128 * 2);         // 64 MB
    f16*   CAT    = (f16*)  alloc((size_t)NIMG * HW * 256 * 2);         // 128 MB (fore|back)

    const int TPB = 256;
    auto cdiv = [](int a, int b) { return (a + b - 1) / b; };

    k_f2h<<<cdiv(128 * 32, TPB),  TPB, 0, stream>>>(sw0, SW0F, 128 * 32);
    k_f2h<<<cdiv(128 * 256, TPB), TPB, 0, stream>>>(ow0, OW0F, 128 * 256);
    k_w3<<<cdiv(128 * 128 * 9, TPB), TPB, 0, stream>>>(sw1, SW1F);
    k_w3<<<cdiv(128 * 128 * 9, TPB), TPB, 0, stream>>>(sw2, SW2F);
    k_w3<<<cdiv(128 * 128 * 9, TPB), TPB, 0, stream>>>(ow1, OW1F);
    k_w3<<<cdiv(128 * 128 * 9, TPB), TPB, 0, stream>>>(ow2, OW2F);

    k_nfs<<<NIMG * HW / TPB, TPB, 0, stream>>>(feats, NFS);
    k_masked<<<NIMG, 256, 0, stream>>>(NFS, sism, MASKED);

    for (int v = 0; v < 2; ++v) {
        const float* mk = MASKED + (size_t)v * NIMG * 256;

        k_proxy<<<4, 256, 0, stream>>>(mk, PROXY, RVEC);
        k_corr<<<NIMG * 256 / 8, TPB, 0, stream>>>(NFS, PROXY, CORR);
        k_cnorm<<<NIMG * 16 * 32 / TPB, TPB, 0, stream>>>(CORR, CNORM);
        k_coatt<<<NIMG, 256, 0, stream>>>(CORR, CNORM, COATT);
        k_score<<<NIMG * HW / TPB, TPB, 0, stream>>>(NFS, RVEC, SCORE);
        k_topk<<<4, 256, 0, stream>>>(SCORE, NFS, COREP);
        k_cofs<<<NIMG * 256 * 2 / 8, TPB, 0, stream>>>(NFS, COREP, COATT, SUBX);
        // Conv2d(32,128,1) + bias -> Y0
        k_conv1x1<32><<<NIMG * 128 * 4 / 8, TPB, 0, stream>>>(SUBX, SW0F, sb0,
                                                              Y0, 128, 0);
        // Res(128): conv3x3+BN+relu -> H1 ; conv3x3 + skip(Y0) + relu -> CAT half
        k_conv3x3<0><<<NIMG * 128 * 4 / 8, TPB, 0, stream>>>(Y0, SW1F, sb1, sg1, sbt1,
                                                             nullptr, H1, nullptr, 128, 0);
        k_conv3x3<1><<<NIMG * 128 * 4 / 8, TPB, 0, stream>>>(H1, SW2F, sb2, nullptr, nullptr,
                                                             Y0, CAT, nullptr, 256, v * 128);
    }

    // outer: Conv2d(256,128,1) on concat + Res(128) -> d_out (f32, NCHW)
    k_conv1x1<256><<<NIMG * 128 * 4 / 8, TPB, 0, stream>>>(CAT, OW0F, ob0, Y0, 128, 0);
    k_conv3x3<0><<<NIMG * 128 * 4 / 8, TPB, 0, stream>>>(Y0, OW1F, ob1, og1, obt1,
                                                         nullptr, H1, nullptr, 128, 0);
    k_conv3x3<2><<<NIMG * 128 * 4 / 8, TPB, 0, stream>>>(H1, OW2F, ob2, nullptr, nullptr,
                                                         Y0, nullptr, out, 0, 0);
}
